// Seq2Seq_40965398069307
// MI455X (gfx1250) — matrix operand order repaired and checked by reading, not compiled, not run on hardware
//
#include <hip/hip_runtime.h>

typedef __attribute__((ext_vector_type(16))) _Float16 v16h;
typedef __attribute__((ext_vector_type(8)))  _Float16 v8h;
typedef __attribute__((ext_vector_type(8)))  float    v8f;

#define BB 16
#define SS 128
#define TT 128
#define EE 512
#define HH 512
#define DD 1024
#define VV 10000
#define VP 10048   // V padded to multiple of 64 (16x64 wave tiles)

__device__ __forceinline__ float sigm(float x) { return 1.0f / (1.0f + __expf(-x)); }

__device__ __forceinline__ v16h mk_a(v8h lo, v8h hi) {
  v16h a;
#pragma unroll
  for (int j = 0; j < 8; ++j) { a[j] = lo[j]; a[j + 8] = hi[j]; }
  return a;
}

// One wave accumulates FOUR 16x16 f32 tiles sharing the same A fragment:
// C[:, 4 tiles of 16] += A[16,K] @ W^T, B tiles = W row blocks at element stride
// `wstride`. Explicitly double-buffered: iteration k+1's fragments are loaded
// before iteration k's WMMAs issue, so loads overlap matrix ops instead of
// serializing on s_wait_loadcnt 0.
__device__ __forceinline__ void wmma_kloop_x4(const _Float16* __restrict__ a_row,
                                              const _Float16* __restrict__ w_row,
                                              long long wstride, int K, int khalf,
                                              int kb16, v8f& c0, v8f& c1, v8f& c2,
                                              v8f& c3) {
  const _Float16* w0 = w_row + khalf;
  const _Float16* w1 = w_row + wstride + khalf;
  const _Float16* w2 = w_row + 2 * wstride + khalf;
  const _Float16* w3 = w_row + 3 * wstride + khalf;
  const _Float16* a0 = a_row + khalf;
  v8h alo = *(const v8h*)(a0);
  v8h ahi = *(const v8h*)(a0 + 16);
  v16h b0 = mk_a(*(const v8h*)(w0), *(const v8h*)(w0 + 16));
  v16h b1 = mk_a(*(const v8h*)(w1), *(const v8h*)(w1 + 16));
  v16h b2 = mk_a(*(const v8h*)(w2), *(const v8h*)(w2 + 16));
  v16h b3 = mk_a(*(const v8h*)(w3), *(const v8h*)(w3 + 16));
  for (int k0 = 32; k0 < K; k0 += 32) {
    __builtin_prefetch(w0 + k0 + 512, 0, 1);  // global_prefetch_b8 (weight stream)
    v8h nalo = *(const v8h*)(a0 + k0);
    v8h nahi = *(const v8h*)(a0 + k0 + 16);
    v16h nb0 = mk_a(*(const v8h*)(w0 + k0), *(const v8h*)(w0 + k0 + 16));
    v16h nb1 = mk_a(*(const v8h*)(w1 + k0), *(const v8h*)(w1 + k0 + 16));
    v16h nb2 = mk_a(*(const v8h*)(w2 + k0), *(const v8h*)(w2 + k0 + 16));
    v16h nb3 = mk_a(*(const v8h*)(w3 + k0), *(const v8h*)(w3 + k0 + 16));
    v16h a = mk_a(alo, ahi);
    c0 = __builtin_amdgcn_wmma_f32_16x16x32_f16(false, a, false, b0, (short)0, c0, false, false);
    c1 = __builtin_amdgcn_wmma_f32_16x16x32_f16(false, a, false, b1, (short)0, c1, false, false);
    c2 = __builtin_amdgcn_wmma_f32_16x16x32_f16(false, a, false, b2, (short)0, c2, false, false);
    c3 = __builtin_amdgcn_wmma_f32_16x16x32_f16(false, a, false, b3, (short)0, c3, false, false);
    alo = nalo; ahi = nahi; b0 = nb0; b1 = nb1; b2 = nb2; b3 = nb3;
  }
  v16h a = mk_a(alo, ahi);
  c0 = __builtin_amdgcn_wmma_f32_16x16x32_f16(false, a, false, b0, (short)0, c0, false, false);
  c1 = __builtin_amdgcn_wmma_f32_16x16x32_f16(false, a, false, b1, (short)0, c1, false, false);
  c2 = __builtin_amdgcn_wmma_f32_16x16x32_f16(false, a, false, b2, (short)0, c2, false, false);
  c3 = __builtin_amdgcn_wmma_f32_16x16x32_f16(false, a, false, b3, (short)0, c3, false, false);
}

// ---------------- utility kernels ----------------
__global__ __launch_bounds__(256) void k_cast(const float* __restrict__ s,
                                              _Float16* __restrict__ d, long long n) {
  long long i = (long long)blockIdx.x * blockDim.x + threadIdx.x;
  long long st = (long long)gridDim.x * blockDim.x;
  for (; i < n; i += st) d[i] = (_Float16)s[i];
}

__global__ __launch_bounds__(256) void k_cast_pad_rows(const float* __restrict__ s,
                                                       _Float16* __restrict__ d,
                                                       int rows, int cols, int rows_pad) {
  long long n = (long long)rows_pad * cols;
  long long st = (long long)gridDim.x * blockDim.x;
  for (long long i = (long long)blockIdx.x * blockDim.x + threadIdx.x; i < n; i += st) {
    int r = (int)(i / cols);
    d[i] = (r < rows) ? (_Float16)s[i] : (_Float16)0.0f;
  }
}

__global__ __launch_bounds__(256) void k_cast_cols(const float* __restrict__ s, int src_ld,
                                                   int col0, _Float16* __restrict__ d,
                                                   int rows, int cols) {
  long long n = (long long)rows * cols;
  long long st = (long long)gridDim.x * blockDim.x;
  for (long long i = (long long)blockIdx.x * blockDim.x + threadIdx.x; i < n; i += st) {
    int r = (int)(i / cols), c = (int)(i % cols);
    d[i] = (_Float16)s[(long long)r * src_ld + col0 + c];
  }
}

__global__ __launch_bounds__(256) void k_pad_bias(const float* __restrict__ s,
                                                  float* __restrict__ d, int n, int np) {
  int i = blockIdx.x * blockDim.x + threadIdx.x;
  if (i < np) d[i] = (i < n) ? s[i] : 0.0f;
}

__global__ __launch_bounds__(256) void k_zero_h(_Float16* p, int n) {
  int i = blockIdx.x * blockDim.x + threadIdx.x;
  if (i < n) p[i] = (_Float16)0.0f;
}
__global__ __launch_bounds__(256) void k_zero_f(float* p, int n) {
  int i = blockIdx.x * blockDim.x + threadIdx.x;
  if (i < n) p[i] = 0.0f;
}

// Embedding gather -> f16, time-major rows (t*B + b). padding_idx==0 -> zeros.
__global__ __launch_bounds__(256) void k_embed(const int* __restrict__ idx,
                                               const float* __restrict__ tab,
                                               _Float16* __restrict__ o,
                                               int Ecols, int Bsz, int seqlen) {
  int r = blockIdx.x;                 // r = t*B + b
  int t = r / Bsz, b = r % Bsz;
  int tok = idx[b * seqlen + t];
  const float* row = tab + (long long)tok * Ecols;
  _Float16* op = o + (long long)r * Ecols;
  for (int e = threadIdx.x; e < Ecols; e += blockDim.x)
    op[e] = (tok == 0) ? (_Float16)0.0f : (_Float16)row[e];
}

// --------- generic WMMA GEMM: C[M,N] = A[M,K] @ W[N,K]^T (+bias), 16x64/wave ---------
__global__ __launch_bounds__(32) void k_gemm(const _Float16* __restrict__ A, int lda,
                                             const _Float16* __restrict__ W, int ldw,
                                             float* __restrict__ C, int ldc, int K,
                                             const float* __restrict__ bias) {
  int lane = threadIdx.x;
  int n0 = blockIdx.x * 64, m0 = blockIdx.y * 16;
  int mr = lane & 15, hi = lane >> 4;
  const _Float16* a_row = A + (long long)(m0 + mr) * lda;
  const _Float16* w_row = W + (long long)(n0 + mr) * ldw;
  v8f c0 = {}, c1 = {}, c2 = {}, c3 = {};
  wmma_kloop_x4(a_row, w_row, (long long)16 * ldw, K, hi * 8, hi * 16, c0, c1, c2, c3);
  int cn = lane & 15, mb = hi * 8;
#pragma unroll
  for (int nt = 0; nt < 4; ++nt) {
    v8f acc = nt == 0 ? c0 : nt == 1 ? c1 : nt == 2 ? c2 : c3;
    int nc = n0 + nt * 16 + cn;
    float bv = bias ? bias[nc] : 0.0f;
#pragma unroll
    for (int r = 0; r < 8; ++r)
      C[(long long)(m0 + mb + r) * ldc + nc] = acc[r] + bv;
  }
}

// ---------------- fused LSTM recurrent step (register-resident pointwise) ----------------
// One WAVE computes all four i/f/g/o 16x16 gate tiles for its 16 hidden columns
// (B row blocks at gate stride Hh*K), sharing the A fragments. Since all four
// accumulators have the identical (m,n)<->(VGPR,lane) mapping, the LSTM pointwise
// update runs entirely in registers: no LDS, no barrier. 4 waves per block.
__global__ __launch_bounds__(128) void k_lstm_step(
    const _Float16* __restrict__ in0, int K0, const _Float16* __restrict__ W0,
    const _Float16* __restrict__ h_prev, const _Float16* __restrict__ Whh, int Hh,
    const float* __restrict__ xg,        // optional precomputed input gates [16,4Hh]
    const float* __restrict__ bias,      // optional [4Hh]
    float* __restrict__ c_state,         // [16,Hh] f32, in-place
    _Float16* __restrict__ c_f16_out,    // optional f16 copy of c
    _Float16* __restrict__ h_out_f16,    // [16,Hh] f16 (ping-pong)
    _Float16* __restrict__ h_arch, long long arch_stride)  // optional archive (offset ptr)
{
  int wid = threadIdx.x >> 5;
  int lane = threadIdx.x & 31;
  int hcol0 = (blockIdx.x * 4 + wid) * 16;
  int mr = lane & 15, hi = lane >> 4;
  v8f gi = {}, gf = {}, gg = {}, go = {};
  if (W0) {
    wmma_kloop_x4(in0 + (long long)mr * K0, W0 + (long long)(hcol0 + mr) * K0,
                  (long long)Hh * K0, K0, hi * 8, hi * 16, gi, gf, gg, go);
  }
  wmma_kloop_x4(h_prev + (long long)mr * Hh, Whh + (long long)(hcol0 + mr) * Hh,
                (long long)Hh * Hh, Hh, hi * 8, hi * 16, gi, gf, gg, go);
  int cn = lane & 15, mb = hi * 8;
  int col = hcol0 + cn;
#pragma unroll
  for (int r = 0; r < 8; ++r) {
    int m = mb + r;
    float vi = gi[r], vf = gf[r], vg = gg[r], vo = go[r];
    if (xg) {
      const float* xr = xg + (long long)m * 4 * Hh;
      vi += xr[col]; vf += xr[Hh + col]; vg += xr[2 * Hh + col]; vo += xr[3 * Hh + col];
    }
    if (bias) {
      vi += bias[col]; vf += bias[Hh + col]; vg += bias[2 * Hh + col]; vo += bias[3 * Hh + col];
    }
    float cp = c_state[(long long)m * Hh + col];
    float cv = sigm(vf) * cp + sigm(vi) * tanhf(vg);
    float hv = sigm(vo) * tanhf(cv);
    c_state[(long long)m * Hh + col] = cv;
    if (c_f16_out) c_f16_out[(long long)m * Hh + col] = (_Float16)cv;
    h_out_f16[(long long)m * Hh + col] = (_Float16)hv;
    if (h_arch) h_arch[(long long)m * arch_stride + col] = (_Float16)hv;
  }
}

// ---------------- attention ----------------
// scores[b,s] = sum_d tanh(enc_proj[b,s,d] + qproj[b,d] + attn_b[d]) * attn_v[d]
__global__ __launch_bounds__(128) void k_attn_scores(const float* __restrict__ enc_proj,
                                                     const float* __restrict__ qproj,
                                                     const float* __restrict__ ab,
                                                     const float* __restrict__ av,
                                                     float* __restrict__ scores) {
  int bs = blockIdx.x;                  // b*S + s
  int b = bs / SS;
  const float* ep = enc_proj + (long long)bs * DD;
  const float* qp = qproj + (long long)b * DD;
  float sum = 0.0f;
  for (int d = threadIdx.x; d < DD; d += 128)
    sum += tanhf(ep[d] + qp[d] + ab[d]) * av[d];
  __shared__ float red[128];
  red[threadIdx.x] = sum;
  __syncthreads();
  for (int o = 64; o; o >>= 1) {
    if (threadIdx.x < o) red[threadIdx.x] += red[threadIdx.x + o];
    __syncthreads();
  }
  if (threadIdx.x == 0) scores[bs] = red[0];
}

// softmax over S and context = sum_s a_s * enc_out[b,s,:], written into dec_in cols [0,2H)
__global__ __launch_bounds__(128) void k_softmax_ctx(const float* __restrict__ scores,
                                                     const _Float16* __restrict__ enc_f16,
                                                     _Float16* __restrict__ ctx, int ctx_ld) {
  int b = blockIdx.x;
  __shared__ float a[SS];
  __shared__ float red[128];
  int tid = threadIdx.x;
  float s = scores[b * SS + tid];
  red[tid] = s;
  __syncthreads();
  for (int o = 64; o; o >>= 1) {
    if (tid < o) red[tid] = fmaxf(red[tid], red[tid + o]);
    __syncthreads();
  }
  float mx = red[0];
  __syncthreads();
  float e = __expf(s - mx);
  red[tid] = e;
  __syncthreads();
  for (int o = 64; o; o >>= 1) {
    if (tid < o) red[tid] += red[tid + o];
    __syncthreads();
  }
  a[tid] = e / red[0];
  __syncthreads();
  const int H2 = 2 * HH;
  for (int c = tid; c < H2; c += 128) {
    float acc = 0.0f;
    for (int si = 0; si < SS; ++si)
      acc += a[si] * (float)enc_f16[((long long)b * SS + si) * H2 + c];
    ctx[(long long)b * ctx_ld + c] = (_Float16)acc;
  }
}

// dec_in cols [2H, 2H+E) = last_emb (zeros at t==0)
__global__ __launch_bounds__(256) void k_set_emb(const _Float16* __restrict__ src,
                                                 _Float16* __restrict__ dec_in, int ld,
                                                 int col0) {
  int i = blockIdx.x * blockDim.x + threadIdx.x;
  if (i >= BB * EE) return;
  int m = i / EE, e = i % EE;
  dec_in[(long long)m * ld + col0 + e] = src ? src[(long long)m * EE + e] : (_Float16)0.0f;
}

__global__ __launch_bounds__(256) void k_store_logits(const float* __restrict__ lg,
                                                      float* __restrict__ out, int t) {
  long long n = (long long)BB * VV;
  long long st = (long long)gridDim.x * blockDim.x;
  for (long long i = (long long)blockIdx.x * blockDim.x + threadIdx.x; i < n; i += st) {
    int b = (int)(i / VV), v = (int)(i % VV);
    out[((long long)b * TT + t) * VV + v] = lg[(long long)b * VP + v];
  }
}

extern "C" void kernel_launch(void* const* d_in, const int* in_sizes, int n_in,
                              void* d_out, int out_size, void* d_ws, size_t ws_size,
                              hipStream_t stream) {
  (void)in_sizes; (void)n_in; (void)out_size; (void)ws_size;
  const int*   x        = (const int*)  d_in[0];
  const int*   gs       = (const int*)  d_in[1];
  const float* in_emb   = (const float*)d_in[2];
  const float* out_emb  = (const float*)d_in[3];
  const float* enc_wih0 = (const float*)d_in[4];
  const float* enc_whh0 = (const float*)d_in[5];
  const float* enc_b0   = (const float*)d_in[6];
  const float* enc_wih1 = (const float*)d_in[7];
  const float* enc_whh1 = (const float*)d_in[8];
  const float* enc_b1   = (const float*)d_in[9];
  const float* dec_wih0 = (const float*)d_in[10];
  const float* dec_whh0 = (const float*)d_in[11];
  const float* dec_b0   = (const float*)d_in[12];
  const float* dec_wih1 = (const float*)d_in[13];
  const float* dec_whh1 = (const float*)d_in[14];
  const float* dec_b1   = (const float*)d_in[15];
  const float* attn_w   = (const float*)d_in[16];
  const float* attn_b   = (const float*)d_in[17];
  const float* attn_v   = (const float*)d_in[18];
  const float* out_w    = (const float*)d_in[19];
  const float* out_b    = (const float*)d_in[20];
  float* out = (float*)d_out;

  const int G4 = 4 * HH;        // 2048 encoder gate width
  const int D4 = 4 * DD;        // 4096 decoder gate width
  const int KD0 = 2 * HH + EE;  // 1536 decoder layer0 input

  // ---- workspace bump allocator (~140 MB total) ----
  char* base = (char*)d_ws; size_t off = 0;
  auto alloc = [&](size_t bytes) -> void* {
    void* p = base + off; off += (bytes + 255) & ~(size_t)255; return p;
  };
  _Float16* wih0h  = (_Float16*)alloc((size_t)2 * G4 * EE * 2);
  _Float16* whh0h  = (_Float16*)alloc((size_t)2 * G4 * HH * 2);
  _Float16* wih1h  = (_Float16*)alloc((size_t)2 * G4 * 2 * HH * 2);
  _Float16* whh1h  = (_Float16*)alloc((size_t)2 * G4 * HH * 2);
  _Float16* dwih0h = (_Float16*)alloc((size_t)D4 * KD0 * 2);
  _Float16* dwhh0h = (_Float16*)alloc((size_t)D4 * DD * 2);
  _Float16* dwih1h = (_Float16*)alloc((size_t)D4 * DD * 2);
  _Float16* dwhh1h = (_Float16*)alloc((size_t)D4 * DD * 2);
  _Float16* awq    = (_Float16*)alloc((size_t)DD * DD * 2);
  _Float16* awe    = (_Float16*)alloc((size_t)DD * 2 * HH * 2);
  _Float16* outwh  = (_Float16*)alloc((size_t)VP * DD * 2);
  float*    outbp  = (float*)   alloc((size_t)VP * 4);
  _Float16* xsh    = (_Float16*)alloc((size_t)SS * BB * EE * 2);
  _Float16* gsh    = (_Float16*)alloc((size_t)TT * BB * EE * 2);
  float*    xgA    = (float*)   alloc((size_t)SS * BB * G4 * 4);
  float*    xgB    = (float*)   alloc((size_t)SS * BB * G4 * 4);
  _Float16* l0h    = (_Float16*)alloc((size_t)SS * BB * 2 * HH * 2);
  _Float16* ench   = (_Float16*)alloc((size_t)BB * SS * 2 * HH * 2);
  float*    eproj  = (float*)   alloc((size_t)BB * SS * DD * 4);
  _Float16* ehA    = (_Float16*)alloc((size_t)BB * HH * 2);
  _Float16* ehB    = (_Float16*)alloc((size_t)BB * HH * 2);
  float*    ec     = (float*)   alloc((size_t)BB * HH * 4);
  _Float16* h0a    = (_Float16*)alloc((size_t)BB * DD * 2);
  _Float16* h0b    = (_Float16*)alloc((size_t)BB * DD * 2);
  _Float16* h1a    = (_Float16*)alloc((size_t)BB * DD * 2);
  _Float16* h1b    = (_Float16*)alloc((size_t)BB * DD * 2);
  float*    c0     = (float*)   alloc((size_t)BB * DD * 4);
  float*    c1     = (float*)   alloc((size_t)BB * DD * 4);
  _Float16* c1h    = (_Float16*)alloc((size_t)BB * DD * 2);
  _Float16* dec_in = (_Float16*)alloc((size_t)BB * KD0 * 2);
  float*    qproj  = (float*)   alloc((size_t)BB * DD * 4);
  float*    scores = (float*)   alloc((size_t)BB * SS * 4);
  float*    logits = (float*)   alloc((size_t)BB * VP * 4);

  // ---- weight conversion fp32 -> f16 (weights then live in L2 for the whole run) ----
#define CAST(s, d, n) k_cast<<<2048, 256, 0, stream>>>((s), (d), (long long)(n))
  CAST(enc_wih0, wih0h, 2 * G4 * EE);
  CAST(enc_whh0, whh0h, 2 * G4 * HH);
  CAST(enc_wih1, wih1h, 2 * G4 * 2 * HH);
  CAST(enc_whh1, whh1h, 2 * G4 * HH);
  CAST(dec_wih0, dwih0h, (size_t)D4 * KD0);
  CAST(dec_whh0, dwhh0h, (size_t)D4 * DD);
  CAST(dec_wih1, dwih1h, (size_t)D4 * DD);
  CAST(dec_whh1, dwhh1h, (size_t)D4 * DD);
#undef CAST
  k_cast_cols<<<2048, 256, 0, stream>>>(attn_w, 2 * HH + DD, 0, awq, DD, DD);
  k_cast_cols<<<2048, 256, 0, stream>>>(attn_w, 2 * HH + DD, DD, awe, DD, 2 * HH);
  k_cast_pad_rows<<<4096, 256, 0, stream>>>(out_w, outwh, VV, DD, VP);
  k_pad_bias<<<(VP + 255) / 256, 256, 0, stream>>>(out_b, outbp, VV, VP);

  // ---- embeddings (padding_idx=0 -> zeros) ----
  k_embed<<<SS * BB, 256, 0, stream>>>(x, in_emb, xsh, EE, BB, SS);
  k_embed<<<TT * BB, 256, 0, stream>>>(gs, out_emb, gsh, EE, BB, TT);

  // ================= encoder =================
  // layer 0: batched input GEMM across all timesteps (bias folded), then recurrent scan
  for (int d = 0; d < 2; ++d) {
    float* xg = d ? xgB : xgA;
    k_gemm<<<dim3(G4 / 64, SS * BB / 16), 32, 0, stream>>>(
        xsh, EE, wih0h + (size_t)d * G4 * EE, EE, xg, G4, EE, enc_b0 + d * G4);
  }
  for (int d = 0; d < 2; ++d) {
    float* xg = d ? xgB : xgA;
    const _Float16* whh = whh0h + (size_t)d * G4 * HH;
    k_zero_h<<<(BB * HH + 255) / 256, 256, 0, stream>>>(ehA, BB * HH);
    k_zero_f<<<(BB * HH + 255) / 256, 256, 0, stream>>>(ec, BB * HH);
    int p = 0;
    for (int st = 0; st < SS; ++st) {
      int t = d ? (SS - 1 - st) : st;
      _Float16* hp = p ? ehB : ehA;
      _Float16* hn = p ? ehA : ehB;
      k_lstm_step<<<HH / 64, 128, 0, stream>>>(
          nullptr, 0, nullptr, hp, whh, HH, xg + (size_t)t * BB * G4, nullptr,
          ec, nullptr, hn, l0h + (size_t)t * BB * 2 * HH + d * HH, (long long)2 * HH);
      p ^= 1;
    }
  }
  // layer 1: batched input GEMM on l0 output, then scan; archive in [b][s] layout
  for (int d = 0; d < 2; ++d) {
    float* xg = d ? xgB : xgA;
    k_gemm<<<dim3(G4 / 64, SS * BB / 16), 32, 0, stream>>>(
        l0h, 2 * HH, wih1h + (size_t)d * G4 * 2 * HH, 2 * HH, xg, G4, 2 * HH,
        enc_b1 + d * G4);
  }
  for (int d = 0; d < 2; ++d) {
    float* xg = d ? xgB : xgA;
    const _Float16* whh = whh1h + (size_t)d * G4 * HH;
    k_zero_h<<<(BB * HH + 255) / 256, 256, 0, stream>>>(ehA, BB * HH);
    k_zero_f<<<(BB * HH + 255) / 256, 256, 0, stream>>>(ec, BB * HH);
    int p = 0;
    for (int st = 0; st < SS; ++st) {
      int t = d ? (SS - 1 - st) : st;
      _Float16* hp = p ? ehB : ehA;
      _Float16* hn = p ? ehA : ehB;
      k_lstm_step<<<HH / 64, 128, 0, stream>>>(
          nullptr, 0, nullptr, hp, whh, HH, xg + (size_t)t * BB * G4, nullptr,
          ec, nullptr, hn, ench + (size_t)t * 2 * HH + d * HH, (long long)SS * 2 * HH);
      p ^= 1;
    }
  }
  // step-invariant attention projection: enc_proj[B*S, D] = enc_out @ W_enc^T
  k_gemm<<<dim3(DD / 64, BB * SS / 16), 32, 0, stream>>>(
      ench, 2 * HH, awe, 2 * HH, eproj, DD, 2 * HH, nullptr);

  // ================= decoder =================
  int p0 = 0, p1 = 0;
  _Float16* h0bufs[2] = {h0a, h0b};
  _Float16* h1bufs[2] = {h1a, h1b};
  k_zero_h<<<(BB * DD + 255) / 256, 256, 0, stream>>>(h0a, BB * DD);
  k_zero_h<<<(BB * DD + 255) / 256, 256, 0, stream>>>(h1a, BB * DD);
  k_zero_f<<<(BB * DD + 255) / 256, 256, 0, stream>>>(c0, BB * DD);
  k_zero_f<<<(BB * DD + 255) / 256, 256, 0, stream>>>(c1, BB * DD);
  k_zero_h<<<(BB * KD0 + 255) / 256, 256, 0, stream>>>(dec_in, BB * KD0);
  // priming step: one dec_step from zero state with zero input
  k_lstm_step<<<DD / 64, 128, 0, stream>>>(dec_in, KD0, dwih0h, h0bufs[p0], dwhh0h, DD,
                                           nullptr, dec_b0, c0, nullptr, h0bufs[p0 ^ 1],
                                           nullptr, 0);
  p0 ^= 1;
  k_lstm_step<<<DD / 64, 128, 0, stream>>>(h0bufs[p0], DD, dwih1h, h1bufs[p1], dwhh1h, DD,
                                           nullptr, dec_b1, c1, c1h, h1bufs[p1 ^ 1],
                                           nullptr, 0);
  p1 ^= 1;

  for (int t = 0; t < TT; ++t) {
    // attention query from last-layer CELL state (reference quirk): qproj = c1 @ W_q^T
    k_gemm<<<dim3(DD / 64, 1), 32, 0, stream>>>(c1h, DD, awq, DD, qproj, DD, DD, nullptr);
    k_attn_scores<<<BB * SS, 128, 0, stream>>>(eproj, qproj, attn_b, attn_v, scores);
    k_softmax_ctx<<<BB, 128, 0, stream>>>(scores, ench, dec_in, KD0);
    k_set_emb<<<(BB * EE + 255) / 256, 256, 0, stream>>>(
        t ? gsh + (size_t)(t - 1) * BB * EE : nullptr, dec_in, KD0, 2 * HH);
    k_lstm_step<<<DD / 64, 128, 0, stream>>>(dec_in, KD0, dwih0h, h0bufs[p0], dwhh0h, DD,
                                             nullptr, dec_b0, c0, nullptr, h0bufs[p0 ^ 1],
                                             nullptr, 0);
    p0 ^= 1;
    k_lstm_step<<<DD / 64, 128, 0, stream>>>(h0bufs[p0], DD, dwih1h, h1bufs[p1], dwhh1h,
                                             DD, nullptr, dec_b1, c1, c1h, h1bufs[p1 ^ 1],
                                             nullptr, 0);
    p1 ^= 1;
    // logits = h_top @ out_w^T + out_b (padded N=10048)
    k_gemm<<<dim3(VP / 64, 1), 32, 0, stream>>>(h1bufs[p1], DD, outwh, DD, logits, VP, DD,
                                                outbp);
    k_store_logits<<<1024, 256, 0, stream>>>(logits, out, t);
  }
}
